// DilatedRNNStack_8229157339297
// MI455X (gfx1250) — compile-verified, matrix operand order checked by reading, not run on hardware
//
#include <hip/hip_runtime.h>

// ---------------------------------------------------------------------------
// Dilated S2-RNN stack for MI455X (gfx1250, wave32, WMMA).
//  - bf16 v_wmma_f32_16x16x32_bf16, fp32 accumulate, fp32 cell state.
//  - Persistent cooperative kernel: 32 WGs x 8 wave32 = 256 waves
//    == 32 M-tiles x 8 S-tiles jobs per layer (each job = 4 gate tiles).
//  - WG owns a fixed S-tile -> its weight slice (76 KB) is staged in LDS once;
//    inner-loop B fragments come from ds_load_b128, A fragments from L2.
//  - Ring buffers with d+1 slots -> write slot never aliases read slots.
//  - Grid barrier: monotonic atomic counter + s_sleep backoff (5 per step).
// ---------------------------------------------------------------------------

#define T_STEPS 512
#define BATCH   512
#define IN_SZ   64
#define H_SZ    32
#define STATE   128
#define OUT_SZ  96
#define OUT_Y   64

#define SLOTS0 2   // d=1  -> d+1
#define SLOTS1 4   // d=3
#define SLOTS2 7   // d=6
#define SLOTS3 13  // d=12
#define HOFF0 0
#define HOFF1 ((size_t)(SLOTS0)*BATCH*H_SZ)
#define HOFF2 ((size_t)(SLOTS0+SLOTS1)*BATCH*H_SZ)
#define HOFF3 ((size_t)(SLOTS0+SLOTS1+SLOTS2)*BATCH*H_SZ)
#define HTOT  ((size_t)(SLOTS0+SLOTS1+SLOTS2+SLOTS3)*BATCH*H_SZ)
#define COFF0 0
#define COFF1 ((size_t)(SLOTS0)*BATCH*STATE)
#define COFF2 ((size_t)(SLOTS0+SLOTS1)*BATCH*STATE)
#define COFF3 ((size_t)(SLOTS0+SLOTS1+SLOTS2)*BATCH*STATE)
#define CTOT  ((size_t)(SLOTS0+SLOTS1+SLOTS2+SLOTS3)*BATCH*STATE)

// LDS weight slice layout (ushort units): per layer, 4 gates x KT tiles x 512
#define LW0_OFF 0
#define LW1_OFF (LW0_OFF + 4*4*512)
#define LW2_OFF (LW1_OFF + 4*5*512)
#define LW3_OFF (LW2_OFF + 4*5*512)
#define LDS_W_TOT (LW3_OFF + 4*5*512)    // 38912 ushorts = 77824 bytes

typedef __attribute__((ext_vector_type(16))) __bf16   bf16x16;
typedef __attribute__((ext_vector_type(8)))  float    f32x8;
typedef __attribute__((ext_vector_type(4)))  float    f32x4;
typedef __attribute__((ext_vector_type(4)))  unsigned u32x4;

union FragU { bf16x16 v; u32x4 q[2]; };

// ---- forced global (address_space(1)) accessors: flat addr bits == global --
#define AS1 __attribute__((address_space(1)))
__device__ __forceinline__ const AS1 u32x4* gp4(const void* p) {
  return (const AS1 u32x4*)(unsigned long long)p;
}
__device__ __forceinline__ float gldf(const float* p) {
  return *(const AS1 float*)(unsigned long long)p;
}
__device__ __forceinline__ void gstf(float* p, float v) {
  *(AS1 float*)(unsigned long long)p = v;
}
__device__ __forceinline__ unsigned short gldu(const unsigned short* p) {
  return *(const AS1 unsigned short*)(unsigned long long)p;
}
__device__ __forceinline__ void gstu(unsigned short* p, unsigned short v) {
  *(AS1 unsigned short*)(unsigned long long)p = v;
}

// A fragment from global: two b128 chunks at +0 and +16 elements
__device__ __forceinline__ bf16x16 load_a_g(const unsigned short* p) {
  FragU u;
  u.q[0] = *gp4(p);
  u.q[1] = *gp4(p + 16);
  return u.v;
}
// B fragment (packed, contiguous 16 elems/lane): chunks at +0 and +8
__device__ __forceinline__ bf16x16 load_b_g(const unsigned short* p) {
  FragU u;
  u.q[0] = *gp4(p);
  u.q[1] = *gp4(p + 8);
  return u.v;
}
// B fragment from LDS (pointer derives from __shared__; inlining => ds_load)
__device__ __forceinline__ bf16x16 load_b_l(const unsigned short* p) {
  FragU u;
  u.q[0] = *(const u32x4*)p;
  u.q[1] = *(const u32x4*)(p + 8);
  return u.v;
}

__device__ __forceinline__ unsigned short f2bf(float f) {
  unsigned u = __float_as_uint(f);
  u += 0x7FFFu + ((u >> 16) & 1u);          // round-to-nearest-even
  return (unsigned short)(u >> 16);
}
__device__ __forceinline__ float bf2f(unsigned short h) {
  return __uint_as_float(((unsigned)h) << 16);
}
__device__ __forceinline__ float sigmoidf_(float x) {
  return 1.0f / (1.0f + __expf(-x));
}
__device__ __forceinline__ float tanhf_(float x) {
  float xa = fminf(fmaxf(x, -12.0f), 12.0f);
  float e = __expf(2.0f * xa);
  return (e - 1.0f) / (e + 1.0f);
}

// ----------------------------- grid barrier --------------------------------
__device__ __forceinline__ void grid_sync(unsigned* bar, int nwg, unsigned* epoch) {
  __threadfence();
  __syncthreads();
  if (threadIdx.x == 0) {
    __hip_atomic_fetch_add(bar, 1u, __ATOMIC_RELEASE, __HIP_MEMORY_SCOPE_AGENT);
    unsigned tgt = *epoch + (unsigned)nwg;
    while (__hip_atomic_load(bar, __ATOMIC_ACQUIRE, __HIP_MEMORY_SCOPE_AGENT) < tgt)
      __builtin_amdgcn_s_sleep(8);
  }
  __syncthreads();
  *epoch += (unsigned)nwg;
  __threadfence();
}

// ------------------------- LDS weight staging ------------------------------
// WG owns S-tile `st`; copy tiles (nt = g*8+st, kt) into LDS as [g][kt][512].
__device__ __forceinline__ void stage_layer(const unsigned short* __restrict__ wp,
                                            unsigned short* lws, int KT, int st) {
  const int total8 = 4 * KT * 64;           // 16-byte chunks
  for (int i = threadIdx.x; i < total8; i += 256) {
    const int tile  = i >> 6;               // g*KT + kt
    const int chunk = i & 63;
    const int g  = tile / KT;
    const int kt = tile - g * KT;
    const int nt = g * 8 + st;
    const u32x4 v = *gp4(wp + ((size_t)(nt * KT + kt) * 512) + chunk * 8);
    *(u32x4*)(lws + (size_t)tile * 512 + chunk * 8) = v;   // ds_store_b128
  }
}

// ------------------------------ layer pass ---------------------------------
template <int IN_KT, bool RES>
__device__ __forceinline__ void layer_pass(
    int mt, int st, int lane,
    const unsigned short* __restrict__ in, int in_ld,
    const unsigned short* lwp,              // LDS weight slice for this layer
    f32x4 bv,                               // per-lane bias for the 4 gates
    unsigned short* __restrict__ hb, float* __restrict__ cb,
    int t, int d,
    unsigned short* __restrict__ outp,
    const unsigned short* __restrict__ resp) {
  constexpr int KT = IN_KT + 2;
  const int col = lane & 15, kh = lane >> 4;
  const int slots = d + 1;
  const int wslot = t % slots;
  const int pslot = (t + slots - 1) % slots;         // state(t-1); zeros at t==0
  const bool hasdel = (t >= d);
  const int dslot = hasdel ? ((t - d) % slots) : pslot;
  const bool hasprev = (t >= 1);

  const unsigned short* hprev = hb + (size_t)pslot * BATCH * H_SZ;
  const unsigned short* hdel  = hb + (size_t)dslot * BATCH * H_SZ;

  f32x8 acc[4];
#pragma unroll
  for (int g = 0; g < 4; ++g)
    acc[g] = (f32x8){bv[g], bv[g], bv[g], bv[g], bv[g], bv[g], bv[g], bv[g]};

  const int row = mt * 16 + col;
#pragma unroll
  for (int kt = 0; kt < KT; ++kt) {
    const unsigned short* src;
    int ld, koff;
    if (kt < IN_KT)       { src = in;    ld = in_ld; koff = kt * 32; }
    else if (kt == IN_KT) { src = hprev; ld = H_SZ;  koff = 0; }
    else                  { src = hdel;  ld = H_SZ;  koff = 0; }
    bf16x16 a = load_a_g(src + (size_t)row * ld + koff + 8 * kh);
#pragma unroll
    for (int g = 0; g < 4; ++g) {
      bf16x16 b = load_b_l(lwp + (size_t)(g * KT + kt) * 512 + lane * 16);
      acc[g] = __builtin_amdgcn_wmma_f32_16x16x32_bf16(
          false, a, false, b, (short)0, acc[g], false, false);
    }
  }

  // ---- gates: lane holds column s for rows m = kh*8 + r (C/D layout) ----
  const int s = st * 16 + col;
  float*       cw = cb + (size_t)wslot * BATCH * STATE;
  const float* cp = cb + (size_t)pslot * BATCH * STATE;
  const float* cd = cb + (size_t)dslot * BATCH * STATE;
  unsigned short* hw = hb + (size_t)wslot * BATCH * H_SZ;

#pragma unroll
  for (int r = 0; r < 8; ++r) {
    const int b = mt * 16 + kh * 8 + r;
    float f  = sigmoidf_(acc[0][r] + 1.0f);
    float cn = tanhf_(acc[1][r]);
    float al = sigmoidf_(acc[2][r]);
    float og = sigmoidf_(acc[3][r]);
    float pc = gldf(cp + (size_t)b * STATE + s);
    float wgt = pc;
    if (hasdel) {
      float dc = gldf(cd + (size_t)b * STATE + s);
      wgt = al * pc + (1.0f - al) * dc;
    }
    float nc = hasprev ? (f * wgt + (1.0f - f) * cn) : cn;
    gstf(cw + (size_t)b * STATE + s, nc);
    float wh = og * nc;
    if (st < 6) {                        // s in [0,96): layer output
      float v = wh;
      if (RES) v += bf2f(gldu(resp + (size_t)b * OUT_SZ + s));
      gstu(outp + (size_t)b * OUT_SZ + s, f2bf(v));
    } else {                             // s in [96,128): new hidden
      gstu(hw + (size_t)b * H_SZ + (s - OUT_SZ), f2bf(wh));
    }
  }
}

// --------------------------- output projection -----------------------------
__device__ __forceinline__ void proj_pass(int pwave, int lane,
                                          const unsigned short* __restrict__ insum,
                                          const unsigned short* __restrict__ wpa,
                                          float bprj, float* __restrict__ y) {
  if (pwave >= 128) return;              // 32 M-tiles x 4 N-tiles
  const int mt = pwave >> 2, nt = pwave & 3;
  const int col = lane & 15, kh = lane >> 4;
  f32x8 acc = (f32x8){bprj, bprj, bprj, bprj, bprj, bprj, bprj, bprj};
#pragma unroll
  for (int kt = 0; kt < 3; ++kt) {       // K = 96
    bf16x16 a = load_a_g(insum + (size_t)(mt * 16 + col) * OUT_SZ + kt * 32 + 8 * kh);
    bf16x16 b = load_b_g(wpa + ((size_t)(nt * 3 + kt) * 32 + lane) * 16);
    acc = __builtin_amdgcn_wmma_f32_16x16x32_bf16(
        false, a, false, b, (short)0, acc, false, false);
  }
#pragma unroll
  for (int r = 0; r < 8; ++r) {
    const int b = mt * 16 + kh * 8 + r;
    gstf(y + (size_t)b * OUT_Y + nt * 16 + col, acc[r]);
  }
}

// ------------------------------ init kernel --------------------------------
__device__ void pack_w(const float* __restrict__ W, unsigned short* __restrict__ wp,
                       int NT, int KT, int K, size_t tid, size_t stride) {
  const size_t total = (size_t)NT * KT * 512;
  for (size_t i = tid; i < total; i += stride) {
    const int j    = (int)(i & 15);
    const int lane = (int)((i >> 4) & 31);
    const size_t tile = i >> 9;
    const int kt = (int)(tile % KT);
    const int nt = (int)(tile / KT);
    const int n = lane & 15, kh = lane >> 4;
    const int kin = (j < 8) ? (8 * kh + j) : (16 + 8 * kh + (j - 8));
    const int k = kt * 32 + kin;
    wp[i] = f2bf(gldf(W + (size_t)(nt * 16 + n) * K + k));
  }
}

__global__ void __launch_bounds__(256) rnn_init(
    const float* __restrict__ x,
    const float* __restrict__ W0, const float* __restrict__ W1,
    const float* __restrict__ W2, const float* __restrict__ W3,
    const float* __restrict__ Wa,
    unsigned short* __restrict__ xbf,
    unsigned short* __restrict__ wp0, unsigned short* __restrict__ wp1,
    unsigned short* __restrict__ wp2, unsigned short* __restrict__ wp3,
    unsigned short* __restrict__ wpa,
    unsigned short* __restrict__ hb, float* __restrict__ cb,
    unsigned* __restrict__ bar) {
  const size_t tid = (size_t)blockIdx.x * blockDim.x + threadIdx.x;
  const size_t stride = (size_t)gridDim.x * blockDim.x;
  if (tid == 0) *bar = 0u;
  for (size_t i = tid; i < (size_t)T_STEPS * BATCH * IN_SZ; i += stride)
    xbf[i] = f2bf(gldf(x + i));
  for (size_t i = tid; i < HTOT; i += stride) hb[i] = 0;
  for (size_t i = tid; i < CTOT; i += stride) cb[i] = 0.0f;
  pack_w(W0, wp0, 32, 4, 128, tid, stride);
  pack_w(W1, wp1, 32, 5, 160, tid, stride);
  pack_w(W2, wp2, 32, 5, 160, tid, stride);
  pack_w(W3, wp3, 32, 5, 160, tid, stride);
  pack_w(Wa, wpa, 4, 3, 96, tid, stride);
}

// --------------------------- persistent kernel -----------------------------
__global__ void __launch_bounds__(256) rnn_persistent(
    const unsigned short* __restrict__ xbf,
    const unsigned short* __restrict__ wp0, const unsigned short* __restrict__ wp1,
    const unsigned short* __restrict__ wp2, const unsigned short* __restrict__ wp3,
    const unsigned short* __restrict__ wpa,
    const float* __restrict__ b0, const float* __restrict__ b1,
    const float* __restrict__ b2, const float* __restrict__ b3,
    const float* __restrict__ ba,
    unsigned short* __restrict__ hb, float* __restrict__ cb,
    unsigned short* __restrict__ out0, unsigned short* __restrict__ out1,
    unsigned short* __restrict__ out2, unsigned short* __restrict__ outsum,
    float* __restrict__ y, unsigned* __restrict__ bar, int nwg) {
  __shared__ unsigned short lds_w[LDS_W_TOT];

  const int wavein = (int)(threadIdx.x >> 5);
  const int lane   = (int)(threadIdx.x & 31);
  const int st = (int)(blockIdx.x & 7);               // WG-owned S-tile
  const int mt = (int)(blockIdx.x >> 3) * 8 + wavein; // wave-owned M-tile
  const int col = lane & 15;
  const int sc = st * 16 + col;

  // Stage this WG's weight slice into LDS (76 KB), once.
  stage_layer(wp0, lds_w + LW0_OFF, 4, st);
  stage_layer(wp1, lds_w + LW1_OFF, 5, st);
  stage_layer(wp2, lds_w + LW2_OFF, 5, st);
  stage_layer(wp3, lds_w + LW3_OFF, 5, st);
  __syncthreads();

  // Hoist all bias values into registers.
  const f32x4 bv0 = {gldf(b0 + sc), gldf(b0 + 128 + sc), gldf(b0 + 256 + sc), gldf(b0 + 384 + sc)};
  const f32x4 bv1 = {gldf(b1 + sc), gldf(b1 + 128 + sc), gldf(b1 + 256 + sc), gldf(b1 + 384 + sc)};
  const f32x4 bv2 = {gldf(b2 + sc), gldf(b2 + 128 + sc), gldf(b2 + 256 + sc), gldf(b2 + 384 + sc)};
  const f32x4 bv3 = {gldf(b3 + sc), gldf(b3 + 128 + sc), gldf(b3 + 256 + sc), gldf(b3 + 384 + sc)};
  const int pwave = (int)blockIdx.x * 8 + wavein;     // projection job id
  const float bprj = (pwave < 128) ? gldf(ba + (pwave & 3) * 16 + col) : 0.0f;

  unsigned epoch = 0;
  for (int t = 0; t < T_STEPS; ++t) {
    // block 0
    layer_pass<2, false>(mt, st, lane, xbf + (size_t)t * BATCH * IN_SZ, IN_SZ,
                         lds_w + LW0_OFF, bv0, hb + HOFF0, cb + COFF0, t, 1,
                         out0, nullptr);
    grid_sync(bar, nwg, &epoch);
    layer_pass<3, false>(mt, st, lane, out0, OUT_SZ,
                         lds_w + LW1_OFF, bv1, hb + HOFF1, cb + COFF1, t, 3,
                         out1, nullptr);
    grid_sync(bar, nwg, &epoch);
    // block 1 (resnet add of out1 fused into layer-3 output)
    layer_pass<3, false>(mt, st, lane, out1, OUT_SZ,
                         lds_w + LW2_OFF, bv2, hb + HOFF2, cb + COFF2, t, 6,
                         out2, nullptr);
    grid_sync(bar, nwg, &epoch);
    layer_pass<3, true>(mt, st, lane, out2, OUT_SZ,
                        lds_w + LW3_OFF, bv3, hb + HOFF3, cb + COFF3, t, 12,
                        outsum, out1);
    grid_sync(bar, nwg, &epoch);
    proj_pass(pwave, lane, outsum, wpa, bprj, y + (size_t)t * BATCH * OUT_Y);
    grid_sync(bar, nwg, &epoch);   // protect outsum before next step's layer 3
  }
}

// ------------------------------- launcher ----------------------------------
extern "C" void kernel_launch(void* const* d_in, const int* in_sizes, int n_in,
                              void* d_out, int out_size, void* d_ws, size_t ws_size,
                              hipStream_t stream) {
  (void)in_sizes; (void)n_in; (void)out_size; (void)ws_size;
  const float* x  = (const float*)d_in[0];
  const float* W0 = (const float*)d_in[1];
  const float* b0 = (const float*)d_in[2];
  const float* W1 = (const float*)d_in[3];
  const float* b1 = (const float*)d_in[4];
  const float* W2 = (const float*)d_in[5];
  const float* b2 = (const float*)d_in[6];
  const float* W3 = (const float*)d_in[7];
  const float* b3 = (const float*)d_in[8];
  const float* Wa = (const float*)d_in[9];
  const float* ba = (const float*)d_in[10];
  float* y = (float*)d_out;

  char* ws = (char*)d_ws;
  size_t off = 0;
  auto take = [&](size_t bytes) {
    size_t o = off;
    off += (bytes + 255) & ~(size_t)255;
    return o;
  };
  unsigned short* xbf = (unsigned short*)(ws + take((size_t)T_STEPS * BATCH * IN_SZ * 2));
  unsigned short* wp0 = (unsigned short*)(ws + take((size_t)32 * 4 * 512 * 2));
  unsigned short* wp1 = (unsigned short*)(ws + take((size_t)32 * 5 * 512 * 2));
  unsigned short* wp2 = (unsigned short*)(ws + take((size_t)32 * 5 * 512 * 2));
  unsigned short* wp3 = (unsigned short*)(ws + take((size_t)32 * 5 * 512 * 2));
  unsigned short* wpa = (unsigned short*)(ws + take((size_t)4 * 3 * 512 * 2));
  unsigned short* hb  = (unsigned short*)(ws + take(HTOT * 2));
  float*          cb  = (float*)(ws + take(CTOT * 4));
  unsigned short* o0  = (unsigned short*)(ws + take((size_t)BATCH * OUT_SZ * 2));
  unsigned short* o1  = (unsigned short*)(ws + take((size_t)BATCH * OUT_SZ * 2));
  unsigned short* o2  = (unsigned short*)(ws + take((size_t)BATCH * OUT_SZ * 2));
  unsigned short* osm = (unsigned short*)(ws + take((size_t)BATCH * OUT_SZ * 2));
  unsigned*       bar = (unsigned*)(ws + take(256));

  rnn_init<<<2048, 256, 0, stream>>>(x, W0, W1, W2, W3, Wa, xbf,
                                     wp0, wp1, wp2, wp3, wpa, hb, cb, bar);

  const int NWG = 32;  // 32 WGs x 8 wave32 = 256 waves = jobs per layer
  rnn_persistent<<<NWG, 256, 0, stream>>>(xbf, wp0, wp1, wp2, wp3, wpa,
                                          b0, b1, b2, b3, ba, hb, cb,
                                          o0, o1, o2, osm, y, bar, NWG);
}
